// MinimalRNNFocusedCell_4956392259879
// MI455X (gfx1250) — matrix operand order repaired and checked by reading, not compiled, not run on hardware
//
#include <hip/hip_runtime.h>
#include <hip/hip_bf16.h>

#define B_ 64
#define T_ 256
#define D_ 512
#define U_ 1024

typedef __bf16 bf16;
typedef __attribute__((ext_vector_type(16))) __bf16 v16bf;
typedef __attribute__((ext_vector_type(8)))  __bf16 v8bf;
typedef __attribute__((ext_vector_type(8)))  float  v8f;
typedef __attribute__((ext_vector_type(4)))  float  v4f;
typedef __attribute__((ext_vector_type(4)))  unsigned int v4u;
typedef __attribute__((ext_vector_type(8)))  int v8i;
typedef __attribute__((ext_vector_type(4)))  int v4i;

static __device__ __forceinline__ v8f wmma_bf16(v16bf a, v16bf b, v8f c) {
  // D(f32 16x16) = A(16x32 bf16) * B(32x16 bf16) + C
  return __builtin_amdgcn_wmma_f32_16x16x32_bf16(false, a, false, b, (short)0, c,
                                                 false, false);
}

// ---------------------------------------------------------------------------
// Init: h ping-pong buffers (h0 -> buf0, zero buf1) + barrier counters.
// Runs every launch so graph replays are deterministic.
// ---------------------------------------------------------------------------
__global__ void k_init(const float* __restrict__ h0, bf16* __restrict__ hbuf,
                       int* __restrict__ cnt) {
  int idx = blockIdx.x * blockDim.x + threadIdx.x;
  int total = 2 * B_ * U_;
  for (int i = idx; i < total; i += gridDim.x * blockDim.x) {
    float v = (i < B_ * U_) ? h0[i] : 0.0f;
    hbuf[i] = (bf16)v;
  }
  if (idx < 4) cnt[idx] = 0;
}

// ---------------------------------------------------------------------------
// Build masked + transposed bf16 weights.
//   block b < 1024 : unit u=b of kernel_current  -> WcT[u*D + j] (n_idx = D)
//   block b >=1024 : unit u of kernel_previous   -> WpT[u*U + j] (n_idx = U)
// mask[u,j] = exp(-(j/(n-1)-mu)^2/(2 si^2)) / ||.||_2 * sqrt(512)
// ---------------------------------------------------------------------------
__global__ void k_masks(const float* __restrict__ Wc, const float* __restrict__ Wp,
                        const float* __restrict__ muc, const float* __restrict__ sic,
                        const float* __restrict__ mup, const float* __restrict__ sip,
                        bf16* __restrict__ WcT, bf16* __restrict__ WpT) {
  __shared__ float red[256];
  int u = blockIdx.x & (U_ - 1);
  bool isPrev = blockIdx.x >= U_;
  int n = isPrev ? U_ : D_;
  float mu = isPrev ? mup[u] : muc[u];
  float si = isPrev ? sip[u] : sic[u];
  si = fminf(fmaxf(si, 0.01f), 1.0f);
  float inv2s2 = 1.0f / (2.0f * si * si);
  float step = 1.0f / (float)(n - 1);

  float ss = 0.0f;
  for (int j = threadIdx.x; j < n; j += 256) {
    float d = j * step - mu;
    float e = __expf(-d * d * inv2s2);
    ss += e * e;
  }
  red[threadIdx.x] = ss;
  __syncthreads();
  for (int s = 128; s > 0; s >>= 1) {
    if ((int)threadIdx.x < s) red[threadIdx.x] += red[threadIdx.x + s];
    __syncthreads();
  }
  float scale = 22.627416997969522f * rsqrtf(red[0]);  // sqrt(512)/||e||

  if (isPrev) {
    for (int j = threadIdx.x; j < n; j += 256) {
      float d = j * step - mu;
      float e = __expf(-d * d * inv2s2);
      WpT[(size_t)u * U_ + j] = (bf16)(Wp[(size_t)j * U_ + u] * e * scale);
    }
  } else {
    for (int j = threadIdx.x; j < n; j += 256) {
      float d = j * step - mu;
      float e = __expf(-d * d * inv2s2);
      WcT[(size_t)u * D_ + j] = (bf16)(Wc[(size_t)j * U_ + u] * e * scale);
    }
  }
}

// ---------------------------------------------------------------------------
// xk = x @ kernel_current  -> written straight into d_out (f32).
// x viewed as [BT=16384, D=512]; one 16x16 tile per wave, K = 512.
// 8 waves / 256-thread block, 8192 blocks. HBM-bound (33 MB in, 67 MB out).
// ---------------------------------------------------------------------------
__global__ void __launch_bounds__(256) k_xk(const float* __restrict__ X,
                                            const bf16* __restrict__ WcT,
                                            float* __restrict__ Out) {
  int lane = threadIdx.x & 31;
  int wv = threadIdx.x >> 5;
  int g = blockIdx.x * 8 + wv;      // wave-tile id, 0..65535
  int mtile = g >> 6;               // 1024 M tiles
  int ntile = g & 63;               // 64 N tiles
  int r0 = mtile * 16;
  int n0 = ntile * 16;
  int lrow = lane & 15;             // A row / B col / C col
  int hi = lane >> 4;               // half-wave select

  const float* arow = X + (size_t)(r0 + lrow) * D_ + hi * 16;
  const bf16* bcol = WcT + (size_t)(n0 + lrow) * D_ + hi * 16;

  v8f acc = {};
  for (int k0 = 0; k0 < D_; k0 += 32) {
    // A fragment: f32 -> bf16 on the fly. lanes<16: K {0..15}, lanes>=16: +16
    v4f x0 = *(const v4f*)(arow + k0);
    v4f x1 = *(const v4f*)(arow + k0 + 4);
    v4f x2 = *(const v4f*)(arow + k0 + 8);
    v4f x3 = *(const v4f*)(arow + k0 + 12);
    v16bf a;
#pragma unroll
    for (int i = 0; i < 4; ++i) {
      a[i]      = (bf16)x0[i];
      a[4 + i]  = (bf16)x1[i];
      a[8 + i]  = (bf16)x2[i];
      a[12 + i] = (bf16)x3[i];
    }
    v16bf b = *(const v16bf*)(bcol + k0);   // 16 contiguous K of column n0+lrow
    acc = wmma_bf16(a, b, acc);
  }
#pragma unroll
  for (int i = 0; i < 8; ++i) {
    int m = i + hi * 8;
    Out[(size_t)(r0 + m) * U_ + (n0 + lrow)] = acc[i];
  }
}

// ---------------------------------------------------------------------------
// Recurrence: h_t = xk_t + h_{t-1} @ Wp, 256 sequential steps.
// 64 WGs of 128 threads (4 waves): blockIdx = btile(4) x ngrp(16).
// Each WG owns rows [16*btile,+16) x cols [64*ngrp,+64); its Wp column slice
// (64 x 1024 bf16 = 128 KB) is staged once into LDS via the Tensor Data
// Mover and reused for all 256 steps. h ping-pongs through a bf16 buffer in
// L2 (32 KB/group/step broadcast); 16-WG atomic barrier per row-group per
// step. d_out is updated in place (xk -> h_t).
// ---------------------------------------------------------------------------
__global__ void __launch_bounds__(128, 1) k_rnn(const bf16* __restrict__ WpT,
                                                float* __restrict__ Out,
                                                bf16* __restrict__ hbuf,
                                                int* __restrict__ barrier_cnt) {
  __shared__ __align__(16) bf16 Bs[64 * U_];   // 128 KB
  int btile = blockIdx.x >> 4;   // 0..3
  int ngrp = blockIdx.x & 15;    // 0..15
  int b0 = btile * 16;
  int n0 = ngrp * 64;
  int lane = threadIdx.x & 31;
  int wv = threadIdx.x >> 5;     // 0..3 -> 16-col subtile

  // ---- Stage this WG's Wp slice (contiguous 128 KB of [N,K]-major WpT) ----
#if __has_builtin(__builtin_amdgcn_tensor_load_to_lds) && \
    __has_builtin(__builtin_amdgcn_s_wait_tensorcnt)
  if (threadIdx.x < 32) {
    // Tensor DMA descriptor (D#), 1-D tile: 8192 x 8B = 64 KB.
    unsigned long long ga =
        (unsigned long long)(const void*)(WpT + (size_t)n0 * U_);
    unsigned int ldsoff = (unsigned int)(unsigned long long)(void*)Bs;
    v4u g0;
    g0[0] = 1u;                                  // count=1, user mode
    g0[1] = ldsoff;                              // lds_addr
    g0[2] = (unsigned int)ga;                    // global_addr[31:0]
    g0[3] = (unsigned int)((ga >> 32) & 0x01FFFFFFull) | (2u << 30);  // type=2
    v8i g1;
    g1[0] = 0x00030000;        // workgroup_mask=0, data_size=3 (8B)
    g1[1] = (int)(8192u << 16);// tensor_dim0 = 8192 (lo16 in [31:16])
    g1[2] = (int)(1u << 16);   // tensor_dim0 hi=0, tensor_dim1 = 1 (lo16)
    g1[3] = (int)(8192u << 16);// tensor_dim1 hi=0, tile_dim0 = 8192
    g1[4] = 0;                 // tile_dim1 = 0 (unused), tile_dim2 = 0
    g1[5] = 8192;              // tensor_dim0_stride lo32
    g1[6] = 0;                 // tensor_dim0_stride hi, tensor_dim1_stride lo
    g1[7] = 0;
    v4i gz4 = {0, 0, 0, 0};
    v8i gz8 = {0, 0, 0, 0, 0, 0, 0, 0};
    __builtin_amdgcn_tensor_load_to_lds(g0, g1, gz4, gz4, gz8, 0);
    __builtin_amdgcn_s_wait_tensorcnt(0);
  }
#else
  {
    const v4f* src = (const v4f*)(WpT + (size_t)n0 * U_);
    v4f* dst = (v4f*)Bs;
    for (int i = threadIdx.x; i < 8192; i += 128) dst[i] = src[i];
  }
#endif
  __syncthreads();

  int lrow = lane & 15;
  int hi = lane >> 4;
  int wcol0 = n0 + wv * 16;
  const bf16* bbase = Bs + (size_t)(wv * 16 + lrow) * U_ + hi * 16;
  int* cnt = barrier_cnt + btile;

  for (int t = 0; t < T_; ++t) {
    const bf16* cur = hbuf + (size_t)(t & 1) * (B_ * U_);
    bf16* nxt = hbuf + (size_t)((t + 1) & 1) * (B_ * U_);
    const bf16* abase = cur + (size_t)(b0 + lrow) * U_ + hi * 16;

    v8f acc = {};
    for (int k0 = 0; k0 < U_; k0 += 32) {
      v8bf a0 = *(const v8bf*)(abase + k0);        // K {0..7}  (+16 for hi lanes)
      v8bf a1 = *(const v8bf*)(abase + k0 + 8);    // K {8..15} (+16 for hi lanes)
      v16bf a = __builtin_shufflevector(a0, a1, 0, 1, 2, 3, 4, 5, 6, 7,
                                        8, 9, 10, 11, 12, 13, 14, 15);
      v16bf b = *(const v16bf*)(bbase + k0);       // 16 contiguous K from LDS
      acc = wmma_bf16(a, b, acc);
    }

    // h_t = acc + xk (in Out); write back f32 to Out, bf16 to next h buffer.
#pragma unroll
    for (int i = 0; i < 8; ++i) {
      int m = i + hi * 8;
      int r = b0 + m;
      int c = wcol0 + lrow;
      size_t oidx = ((size_t)r * T_ + t) * U_ + c;
      float v = acc[i] + Out[oidx];
      Out[oidx] = v;
      nxt[(size_t)r * U_ + c] = (bf16)v;
    }

    // 16-WG barrier for this row-group (monotonic counter, re-zeroed by k_init).
    __threadfence();
    __syncthreads();
    if (threadIdx.x == 0) {
      __hip_atomic_fetch_add(cnt, 1, __ATOMIC_RELEASE, __HIP_MEMORY_SCOPE_AGENT);
      int target = 16 * (t + 1);
      while (__hip_atomic_load(cnt, __ATOMIC_ACQUIRE, __HIP_MEMORY_SCOPE_AGENT) <
             target) {
        __builtin_amdgcn_s_sleep(1);
      }
    }
    __syncthreads();
  }
}

// ---------------------------------------------------------------------------
extern "C" void kernel_launch(void* const* d_in, const int* in_sizes, int n_in,
                              void* d_out, int out_size, void* d_ws, size_t ws_size,
                              hipStream_t stream) {
  const float* x   = (const float*)d_in[0];
  const float* h0  = (const float*)d_in[1];
  const float* Wc  = (const float*)d_in[2];
  const float* Wp  = (const float*)d_in[3];
  const float* muc = (const float*)d_in[4];
  const float* sic = (const float*)d_in[5];
  const float* mup = (const float*)d_in[6];
  const float* sip = (const float*)d_in[7];
  float* out = (float*)d_out;

  char* ws = (char*)d_ws;
  bf16* WcT  = (bf16*)ws;                                   // 1 MB  [U, D]
  bf16* WpT  = (bf16*)(ws + (1u << 20));                    // 2 MB  [U, U] (N-major)
  bf16* hbuf = (bf16*)(ws + (3u << 20));                    // 256 KB ping-pong h
  int* cnt   = (int*)(ws + (3u << 20) + (1u << 18));        // 4 barrier counters

  hipLaunchKernelGGL(k_init, dim3(64), dim3(256), 0, stream, h0, hbuf, cnt);
  hipLaunchKernelGGL(k_masks, dim3(2 * U_), dim3(256), 0, stream,
                     Wc, Wp, muc, sic, mup, sip, WcT, WpT);
  hipLaunchKernelGGL(k_xk, dim3(8192), dim3(256), 0, stream, x, WcT, out);
  hipLaunchKernelGGL(k_rnn, dim3(64), dim3(128), 0, stream, WpT, out, hbuf, cnt);
}
